// MA_CB_RP_Major_60833916781072
// MI455X (gfx1250) — matrix operand order repaired and checked by reading, not compiled, not run on hardware
//
#include <hip/hip_runtime.h>
#include <hip/hip_bf16.h>

#define A_  10
#define B_  32
#define FS_ 2
#define FD_ 8
#define H_  512
#define S_  512

typedef __attribute__((ext_vector_type(16))) __bf16 v16bf;
typedef __attribute__((ext_vector_type(8)))  float  v8f;

__device__ __forceinline__ float sigmoidf_(float x) { return 1.0f / (1.0f + __expf(-x)); }

// Native CDNA5 V_TANH_F32 (single TRANS op, co-executes with VALU) when available.
__device__ __forceinline__ float fast_tanh(float x) {
#if __has_builtin(__builtin_amdgcn_tanh_f32)
    return __builtin_amdgcn_tanh_f32(x);
#elif __has_builtin(__builtin_amdgcn_tanhf)
    return __builtin_amdgcn_tanhf(x);
#else
    return tanhf(x);
#endif
}

// ---------------------------------------------------------------------------
// Weight folding: collapse rank-2/rank-8 encoders through attention/pointer
// weights.  One thread per (a,h), K=512 reduction.
//   UaS[a,h,0:2] = W_attn[a,h,0:H]   @ Ws          (attention, static path)
//   UaD[a,h,0:8] = W_attn[a,h,H:2H]  @ Wd[a]       (attention, dynamic path)
//   abf[a,h]     = W_attn[a,h,0:H]@bs + W_attn[a,h,H:2H]@bd[a]
//   PpS[a,h,0:2] = W_ptr[a,h,0:H]    @ Ws          (pointer, static path)
//   Pc [a,h,0:2] = W_ptr[a,h,H:2H]   @ Ws          (pointer, ctx path; ctx = Ws@cst + bs)
//   c0 [a,h]     = (W_ptr[a,h,0:H] + W_ptr[a,h,H:2H]) @ bs
// ---------------------------------------------------------------------------
__global__ void fold_kernel(const float* __restrict__ Ws, const float* __restrict__ bs,
                            const float* __restrict__ Wd, const float* __restrict__ bd,
                            const float* __restrict__ W_attn, const float* __restrict__ W_ptr,
                            float* __restrict__ UaS, float* __restrict__ UaD,
                            float* __restrict__ abf, float* __restrict__ PpS,
                            float* __restrict__ Pc, float* __restrict__ c0)
{
    int idx = blockIdx.x * blockDim.x + threadIdx.x;   // idx = a*H + h
    if (idx >= A_ * H_) return;
    int a = idx / H_;
    const float* Wa = W_attn + (long)idx * 3 * H_;
    const float* Wp = W_ptr  + (long)idx * 2 * H_;
    float us0 = 0.f, us1 = 0.f, ab = 0.f;
    float ps0 = 0.f, ps1 = 0.f, pc0 = 0.f, pc1 = 0.f, cc = 0.f;
    float ud[FD_];
#pragma unroll
    for (int f = 0; f < FD_; ++f) ud[f] = 0.f;
    for (int k = 0; k < H_; ++k) {
        float was = Wa[k], wad = Wa[H_ + k];
        float w0 = Ws[2 * k], w1 = Ws[2 * k + 1];
        us0 += was * w0;  us1 += was * w1;
        ab  += was * bs[k] + wad * bd[(long)a * H_ + k];
        const float* wdk = Wd + ((long)a * H_ + k) * FD_;
#pragma unroll
        for (int f = 0; f < FD_; ++f) ud[f] += wad * wdk[f];
        float wps = Wp[k], wpc = Wp[H_ + k];
        ps0 += wps * w0;  ps1 += wps * w1;
        pc0 += wpc * w0;  pc1 += wpc * w1;
        cc  += (wps + wpc) * bs[k];
    }
    UaS[2 * idx] = us0;  UaS[2 * idx + 1] = us1;
#pragma unroll
    for (int f = 0; f < FD_; ++f) UaD[(long)idx * FD_ + f] = ud[f];
    abf[idx] = ab;
    PpS[2 * idx] = ps0;  PpS[2 * idx + 1] = ps1;
    Pc [2 * idx] = pc0;  Pc [2 * idx + 1] = pc1;
    c0[idx] = cc;
}

// ---------------------------------------------------------------------------
// x[a,b,h] = Wdec[a,h,:] @ decoder_input[b,:,0] + bdec[a,h]   (K = FS = 2)
// ---------------------------------------------------------------------------
__global__ void x_kernel(const float* __restrict__ Wdec, const float* __restrict__ bdec,
                         const float* __restrict__ dec, float* __restrict__ x)
{
    int idx = blockIdx.x * blockDim.x + threadIdx.x;   // a*B*H
    if (idx >= A_ * B_ * H_) return;
    int h = idx % H_;
    int ab = idx / H_;
    int b = ab % B_;
    int a = ab / B_;
    long wrow = ((long)a * H_ + h) * FS_;
    x[idx] = Wdec[wrow] * dec[b * FS_] + Wdec[wrow + 1] * dec[b * FS_ + 1]
           + bdec[(long)a * H_ + h];
}

// ---------------------------------------------------------------------------
// Generic batched skinny GEMM on the WMMA path:
//   C[z][M,N] = A[z][M,K] * Bw[z][N,K]^T + bias[z][N]
// bf16 inputs (converted in-register from f32), f32 accumulate via
// v_wmma_f32_16x16x32_bf16.  One wave per 16x16 output tile, blockDim=(32,4).
// Fragment layouts used here (wave32; A/B in contiguous k order):
//   A elem j : K = 16*(lane>>4) + j
//   B elem j : K = 16*(lane>>4) + j
//   C vgpr r : M = r + 8*(lane>>4), N = lane&15
// K-loop unrolled x2 -> two independent load->cvt->wmma chains in flight.
// ---------------------------------------------------------------------------
__global__ void gemm_bf16_wmma(const float* __restrict__ A, long sA, int lda,
                               const float* __restrict__ Bw, long sB, int ldb,
                               const float* __restrict__ bias, long sBias,
                               float* __restrict__ C, long sC,
                               int M, int N, int K)
{
    const int z  = blockIdx.z;
    const int n0 = (blockIdx.x * blockDim.y + threadIdx.y) * 16;
    const int m0 = blockIdx.y * 16;
    if (n0 >= N || m0 >= M) return;
    const int lane = threadIdx.x;
    const int half = lane >> 4;
    const int r16  = lane & 15;

    const float* Ap = A  + (long)z * sA + (long)(m0 + r16) * lda;
    const float* Bp = Bw + (long)z * sB + (long)(n0 + r16) * ldb;

    v8f acc = {};
#pragma unroll 2
    for (int k0 = 0; k0 < K; k0 += 32) {
        v16bf fa, fb;
        const float* ar = Ap + k0 + half * 16;
        const float* br = Bp + k0 + half * 16;
        // weights are streamed exactly once: keep HBM ahead of the cvt/wmma chain
        __builtin_prefetch(br + 32, 0, 0);
#pragma unroll
        for (int j = 0; j < 8; ++j) {
            fa[j]     = (__bf16)ar[j];
            fa[j + 8] = (__bf16)ar[8 + j];
        }
#pragma unroll
        for (int j = 0; j < 16; ++j) fb[j] = (__bf16)br[j];
        acc = __builtin_amdgcn_wmma_f32_16x16x32_bf16(false, fa, false, fb,
                                                      (short)0, acc, false, false);
    }
    float* Cp = C + (long)z * sC;
    const float bv = bias[(long)z * sBias + n0 + r16];
#pragma unroll
    for (int r = 0; r < 8; ++r) {
        int mm = m0 + r + 8 * half;
        Cp[(long)mm * N + n0 + r16] = acc[r] + bv;
    }
}

// ---------------------------------------------------------------------------
// GRU gates (torch order r,z,n): h_new = (1-z)*n + z*h_prev
// ---------------------------------------------------------------------------
__global__ void gru_kernel(const float* __restrict__ gi, const float* __restrict__ gh,
                           const float* __restrict__ hprev, float* __restrict__ hh)
{
    int idx = blockIdx.x * blockDim.x + threadIdx.x;   // a*B*H
    if (idx >= A_ * B_ * H_) return;
    int h = idx % H_;
    int ab = idx / H_;
    int b = ab % B_;
    const float* gia = gi + (long)ab * 3 * H_;
    const float* gha = gh + (long)ab * 3 * H_;
    float r  = sigmoidf_(gia[h]          + gha[h]);
    float z  = sigmoidf_(gia[H_ + h]     + gha[H_ + h]);
    float n  = fast_tanh(gia[2 * H_ + h] + r * gha[2 * H_ + h]);
    float hp = hprev[(long)b * H_ + h];
    hh[idx] = (1.f - z) * n + z * hp;
}

// ---------------------------------------------------------------------------
// Attention logits: logit[a,b,s] = sum_h va[h]*tanh(q[a,b,h]
//                         + UaS[h,:]@static[b,:,s] + UaD[h,:]@dyn[a,b,:,s])
// q already contains Wah@h_new + folded biases.  One thread per station.
// ---------------------------------------------------------------------------
__global__ void attn_logits_kernel(const float* __restrict__ st, const float* __restrict__ dyn,
                                   const float* __restrict__ q, const float* __restrict__ UaS,
                                   const float* __restrict__ UaD, const float* __restrict__ va,
                                   float* __restrict__ logits)
{
    __shared__ float sq [H_];
    __shared__ float sva[H_];
    __shared__ float sus[H_ * 2];
    __shared__ float sud[H_ * FD_];
    const int a = blockIdx.z, b = blockIdx.y;
    for (int i = threadIdx.x; i < H_; i += blockDim.x) {
        sq [i] = q [((long)a * B_ + b) * H_ + i];
        sva[i] = va[(long)a * H_ + i];
        sus[2 * i]     = UaS[((long)a * H_ + i) * 2];
        sus[2 * i + 1] = UaS[((long)a * H_ + i) * 2 + 1];
#pragma unroll
        for (int f = 0; f < FD_; ++f)
            sud[i * FD_ + f] = UaD[((long)a * H_ + i) * FD_ + f];
    }
    __syncthreads();
    const int s = blockIdx.x * blockDim.x + threadIdx.x;
    const float st0 = st[((long)b * FS_    ) * S_ + s];
    const float st1 = st[((long)b * FS_ + 1) * S_ + s];
    float dy[FD_];
#pragma unroll
    for (int f = 0; f < FD_; ++f)
        dy[f] = dyn[(((long)a * B_ + b) * FD_ + f) * S_ + s];
    float acc = 0.f;
    for (int h = 0; h < H_; ++h) {
        float v = sq[h] + sus[2 * h] * st0 + sus[2 * h + 1] * st1;
#pragma unroll
        for (int f = 0; f < FD_; ++f) v += sud[h * FD_ + f] * dy[f];
        acc += sva[h] * fast_tanh(v);
    }
    logits[((long)a * B_ + b) * S_ + s] = acc;
}

// ---------------------------------------------------------------------------
// Softmax over S, then cst[f] = sum_s attn[s]*static[b,f,s], then
// r2[a,b,h] = Pc[a,h,:]@cst + c0[a,h]   (pointer-head per-(b,h) constant).
// One block per (a,b), 256 threads, 2 stations each.
// ---------------------------------------------------------------------------
__global__ void softmax_ctx_kernel(const float* __restrict__ logits,
                                   const float* __restrict__ st,
                                   const float* __restrict__ Pc,
                                   const float* __restrict__ c0,
                                   float* __restrict__ r2)
{
    const int b = blockIdx.x, a = blockIdx.y;
    const int t = threadIdx.x;
    __shared__ float red[256];
    const float* L = logits + ((long)a * B_ + b) * S_;
    float l0 = L[t], l1 = L[t + 256];
    red[t] = fmaxf(l0, l1);
    __syncthreads();
    for (int o = 128; o > 0; o >>= 1) { if (t < o) red[t] = fmaxf(red[t], red[t + o]); __syncthreads(); }
    float mx = red[0];
    __syncthreads();
    float e0 = __expf(l0 - mx), e1 = __expf(l1 - mx);
    red[t] = e0 + e1;
    __syncthreads();
    for (int o = 128; o > 0; o >>= 1) { if (t < o) red[t] += red[t + o]; __syncthreads(); }
    float inv = 1.0f / red[0];
    __syncthreads();
    const float* s0 = st + (long)b * FS_ * S_;
    const float* s1 = s0 + S_;
    red[t] = e0 * s0[t] + e1 * s0[t + 256];
    __syncthreads();
    for (int o = 128; o > 0; o >>= 1) { if (t < o) red[t] += red[t + o]; __syncthreads(); }
    float cst0 = red[0] * inv;
    __syncthreads();
    red[t] = e0 * s1[t] + e1 * s1[t + 256];
    __syncthreads();
    for (int o = 128; o > 0; o >>= 1) { if (t < o) red[t] += red[t + o]; __syncthreads(); }
    float cst1 = red[0] * inv;
    __syncthreads();
    for (int h = t; h < H_; h += 256) {
        long w = (long)a * H_ + h;
        r2[((long)a * B_ + b) * H_ + h] = Pc[2 * w] * cst0 + Pc[2 * w + 1] * cst1 + c0[w];
    }
}

// ---------------------------------------------------------------------------
// Pointer logits (final probs output):
//   probs[a,b,s] = sum_h vp[h]*tanh(r2[a,b,h] + PpS[h,:]@static[b,:,s])
// ---------------------------------------------------------------------------
__global__ void ptr_logits_kernel(const float* __restrict__ st, const float* __restrict__ r2,
                                  const float* __restrict__ PpS, const float* __restrict__ vp,
                                  float* __restrict__ probs)
{
    __shared__ float sr [H_];
    __shared__ float svp[H_];
    __shared__ float sps[H_ * 2];
    const int a = blockIdx.z, b = blockIdx.y;
    for (int i = threadIdx.x; i < H_; i += blockDim.x) {
        sr [i] = r2[((long)a * B_ + b) * H_ + i];
        svp[i] = vp[(long)a * H_ + i];
        sps[2 * i]     = PpS[((long)a * H_ + i) * 2];
        sps[2 * i + 1] = PpS[((long)a * H_ + i) * 2 + 1];
    }
    __syncthreads();
    const int s = blockIdx.x * blockDim.x + threadIdx.x;
    const float st0 = st[((long)b * FS_    ) * S_ + s];
    const float st1 = st[((long)b * FS_ + 1) * S_ + s];
    float acc = 0.f;
    for (int h = 0; h < H_; ++h)
        acc += svp[h] * fast_tanh(sr[h] + sps[2 * h] * st0 + sps[2 * h + 1] * st1);
    probs[((long)a * B_ + b) * S_ + s] = acc;
}

// ---------------------------------------------------------------------------
extern "C" void kernel_launch(void* const* d_in, const int* in_sizes, int n_in,
                              void* d_out, int out_size, void* d_ws, size_t ws_size,
                              hipStream_t stream)
{
    (void)in_sizes; (void)n_in; (void)out_size; (void)ws_size;

    const float* st      = (const float*)d_in[0];   // [B,FS,S]
    const float* dyn     = (const float*)d_in[1];   // [A,B,FD,S]
    const float* dec     = (const float*)d_in[2];   // [B,FS,1]
    const float* last_hh = (const float*)d_in[3];   // [B,H]
    const float* Ws      = (const float*)d_in[4];   // [H,FS]
    const float* bs      = (const float*)d_in[5];   // [H]
    const float* Wd      = (const float*)d_in[6];   // [A,H,FD]
    const float* bd      = (const float*)d_in[7];   // [A,H]
    const float* Wdec    = (const float*)d_in[8];   // [A,H,FS]
    const float* bdec    = (const float*)d_in[9];   // [A,H]
    const float* Wih     = (const float*)d_in[10];  // [A,3H,H]
    const float* Whh     = (const float*)d_in[11];  // [A,3H,H]
    const float* bih     = (const float*)d_in[12];  // [A,3H]
    const float* bhh     = (const float*)d_in[13];  // [A,3H]
    const float* v_attn  = (const float*)d_in[14];  // [A,H]
    const float* W_attn  = (const float*)d_in[15];  // [A,H,3H]
    const float* v_ptr   = (const float*)d_in[16];  // [A,H]
    const float* W_ptr   = (const float*)d_in[17];  // [A,H,2H]

    float* probs = (float*)d_out;                    // [A,B,S]
    float* hh    = probs + (long)A_ * B_ * S_;       // [A,B,H]

    // workspace layout (f32)
    float* ws  = (float*)d_ws;
    float* x      = ws;                              // A*B*H
    float* gi     = x      + (long)A_ * B_ * H_;     // A*B*3H
    float* gh     = gi     + (long)A_ * B_ * 3 * H_; // A*B*3H
    float* q      = gh     + (long)A_ * B_ * 3 * H_; // A*B*H
    float* logits = q      + (long)A_ * B_ * H_;     // A*B*S
    float* r2     = logits + (long)A_ * B_ * S_;     // A*B*H
    float* UaS    = r2     + (long)A_ * B_ * H_;     // A*H*2
    float* UaD    = UaS    + (long)A_ * H_ * 2;      // A*H*8
    float* abf    = UaD    + (long)A_ * H_ * FD_;    // A*H
    float* PpS    = abf    + (long)A_ * H_;          // A*H*2
    float* Pc     = PpS    + (long)A_ * H_ * 2;      // A*H*2
    float* c0     = Pc     + (long)A_ * H_ * 2;      // A*H

    // 1) fold weights + decoder-input encoding
    fold_kernel<<<(A_ * H_ + 255) / 256, 256, 0, stream>>>(Ws, bs, Wd, bd, W_attn, W_ptr,
                                                           UaS, UaD, abf, PpS, Pc, c0);
    x_kernel<<<(A_ * B_ * H_ + 255) / 256, 256, 0, stream>>>(Wdec, bdec, dec, x);

    // 2) GRU gate GEMMs (WMMA bf16): gi = x@Wih^T + bih, gh = h@Whh^T + bhh
    dim3 gblk(32, 4, 1);
    gemm_bf16_wmma<<<dim3(24, 2, A_), gblk, 0, stream>>>(
        x, (long)B_ * H_, H_, Wih, (long)3 * H_ * H_, H_, bih, (long)3 * H_,
        gi, (long)B_ * 3 * H_, B_, 3 * H_, H_);
    gemm_bf16_wmma<<<dim3(24, 2, A_), gblk, 0, stream>>>(
        last_hh, 0L, H_, Whh, (long)3 * H_ * H_, H_, bhh, (long)3 * H_,
        gh, (long)B_ * 3 * H_, B_, 3 * H_, H_);

    // 3) GRU elementwise -> h_new (second half of d_out)
    gru_kernel<<<(A_ * B_ * H_ + 255) / 256, 256, 0, stream>>>(gi, gh, last_hh, hh);

    // 4) q = h_new @ Wah^T + abf (WMMA bf16); Wah = W_attn[:, :, 2H:3H] strided
    gemm_bf16_wmma<<<dim3(8, 2, A_), gblk, 0, stream>>>(
        hh, (long)B_ * H_, H_, W_attn + 2 * H_, (long)H_ * 3 * H_, 3 * H_,
        abf, (long)H_, q, (long)B_ * H_, B_, H_, H_);

    // 5) attention logits, 6) softmax + ctx-fold + pointer constants, 7) probs
    attn_logits_kernel<<<dim3(S_ / 256, B_, A_), 256, 0, stream>>>(st, dyn, q, UaS, UaD,
                                                                   v_attn, logits);
    softmax_ctx_kernel<<<dim3(B_, A_), 256, 0, stream>>>(logits, st, Pc, c0, r2);
    ptr_logits_kernel<<<dim3(S_ / 256, B_, A_), 256, 0, stream>>>(st, r2, PpS, v_ptr, probs);
}
